// BinaryTreeLSTM_12180527251606
// MI455X (gfx1250) — matrix operand order repaired and checked by reading, not compiled, not run on hardware
//
#include <hip/hip_runtime.h>
#include <hip/hip_bf16.h>
#include <stdint.h>

// ---------------------------------------------------------------------------
// Binary TreeLSTM for MI455X (gfx1250), wave32, WMMA bf16 path.
// B=64, L=1024, IN_DIM=300, MEM=256.
//
// One-time fp32->bf16 prep (padded -> check-free GEMM loops), then a leaf
// GEMM + 10 tree-level GEMMs through v_wmma_f32_16x16x32_bf16. A tiles are
// staged through a DOUBLE-BUFFERED LDS pipeline fed by CDNA5 async
// global->LDS loads (copy of tile ks+1 overlaps WMMA of tile ks; exactly one
// barrier per K-step). B fragments are direct 32-byte bf16 loads.
// ---------------------------------------------------------------------------

constexpr int kB   = 64;
constexpr int kL   = 1024;
constexpr int kIn  = 300;
constexpr int kInP = 320;    // padded K for the leaf GEMM (10 x 32)
constexpr int kMem = 256;

typedef __attribute__((ext_vector_type(16))) __bf16 v16bf;
typedef __attribute__((ext_vector_type(8)))  __bf16 v8bf;
typedef __attribute__((ext_vector_type(8)))  float  v8f;
typedef int v4i __attribute__((vector_size(16)));   // matches builtin param

__device__ __forceinline__ float sigmoidf_(float x) {
    return 1.0f / (1.0f + __expf(-x));
}
__device__ __forceinline__ float tanhf_(float x) {
    float ax = fabsf(x);
    float e  = __expf(-2.0f * ax);
    float t  = (1.0f - e) / (1.0f + e);
    return (x < 0.0f) ? -t : t;
}

// 16-byte global -> LDS copy: async tensor-class path when available.
__device__ __forceinline__ void async_copy16(const __bf16* g, __bf16* l) {
#if __has_builtin(__builtin_amdgcn_global_load_async_to_lds_b128)
    __builtin_amdgcn_global_load_async_to_lds_b128(
        (__attribute__((address_space(1))) v4i*)(uintptr_t)g,
        (__attribute__((address_space(3))) v4i*)(uintptr_t)l, 0, 0);
#else
    *(v8bf*)l = *(const v8bf*)g;
#endif
}
__device__ __forceinline__ void wait_async0() {
#if __has_builtin(__builtin_amdgcn_s_wait_asynccnt)
    __builtin_amdgcn_s_wait_asynccnt(0);
#elif defined(__gfx1250__)
    asm volatile("s_wait_asynccnt 0" ::: "memory");
#endif
}

// 16x32 bf16 A-fragment from a row-major 16x32 LDS tile. contiguous K order used here (B matches):
// lanes 0-15 -> row=lane, K={0..15}; lanes 16-31 -> K={16..31}.
__device__ __forceinline__ v16bf load_a_frag(const __bf16* As, int rl, int half) {
    const v8bf lo = *(const v8bf*)(As + rl * 32 + half * 16);
    const v8bf hi = *(const v8bf*)(As + rl * 32 + half * 16 + 8);
    return __builtin_shufflevector(lo, hi, 0,1,2,3,4,5,6,7,8,9,10,11,12,13,14,15);
}

// ---------------------------------------------------------------------------
// Prep kernels: fp32 -> bf16 conversions, padded.
// ---------------------------------------------------------------------------
__global__ __launch_bounds__(256) void conv_embs_kernel(
    const float* __restrict__ embs, __bf16* __restrict__ out)
{
    const int idx = blockIdx.x * 256 + threadIdx.x;   // (B*L) * 80 threads
    const int row = idx / (kInP / 4);
    const int k4  = (idx % (kInP / 4)) * 4;
    const float* src = embs + (size_t)row * kIn;
    __bf16* dst = out + (size_t)row * kInP + k4;
    #pragma unroll
    for (int q = 0; q < 4; ++q) {
        const int k = k4 + q;
        dst[q] = (__bf16)((k < kIn) ? src[k] : 0.0f);
    }
}

__global__ __launch_bounds__(256) void conv_wx_kernel(
    const float* __restrict__ Wcx, const float* __restrict__ Wox,
    __bf16* __restrict__ wcp, __bf16* __restrict__ wop)
{
    const int idx = blockIdx.x * 256 + threadIdx.x;   // 256 * 80 threads
    const int row = idx / (kInP / 4);
    const int k4  = (idx % (kInP / 4)) * 4;
    const float* sc = Wcx + (size_t)row * kIn;
    const float* so = Wox + (size_t)row * kIn;
    __bf16* dc  = wcp + (size_t)row * kInP + k4;
    __bf16* dow = wop + (size_t)row * kInP + k4;
    #pragma unroll
    for (int q = 0; q < 4; ++q) {
        const int k = k4 + q;
        dc[q]  = (__bf16)((k < kIn) ? sc[k] : 0.0f);
        dow[q] = (__bf16)((k < kIn) ? so[k] : 0.0f);
    }
}

// Fused [Wl;Wr] bf16 B-matrix: Wb[(g*256+o)*512 + k], k<256 from Wl, else Wr.
__global__ __launch_bounds__(256) void conv_wlr_kernel(
    const float* __restrict__ Wl, const float* __restrict__ Wr,
    __bf16* __restrict__ wb)
{
    const int idx = blockIdx.x * 256 + threadIdx.x;   // 1024 * 128 threads
    const int go  = idx >> 7;                          // g*256 + o
    const int k4  = (idx & 127) * 4;                   // 0..508
    const float* src = ((k4 < kMem) ? Wl : Wr) + (size_t)go * kMem + (k4 & (kMem - 1));
    __bf16* dst = wb + (size_t)go * 512 + k4;
    #pragma unroll
    for (int q = 0; q < 4; ++q) dst[q] = (__bf16)src[q];
}

// ---------------------------------------------------------------------------
// Leaf kernel: block = 256 threads (8 waves), tile 16 rows x 128 cols.
// Each wave: one 16x16 tile for both the c-GEMM and the o-GEMM. K = 320.
// ---------------------------------------------------------------------------
__global__ __launch_bounds__(256) void tlstm_leaf_kernel(
    const __bf16* __restrict__ embs_bf,
    const __bf16* __restrict__ wcp, const __bf16* __restrict__ wop,
    const float* __restrict__ bcx, const float* __restrict__ box,
    float* __restrict__ cbuf, __bf16* __restrict__ hbf)
{
    constexpr int NS = kInP / 32;              // 10 K-steps
    __shared__ __align__(32) __bf16 As[2][16 * 32];

    const int t    = threadIdx.x;
    const int wave = t >> 5;
    const int lane = t & 31;
    const int half = lane >> 4;
    const int rl   = lane & 15;
    const int row0 = blockIdx.x * 16;
    const int col  = blockIdx.y * 128 + wave * 16 + rl;

    const __bf16* wcb = wcp + (size_t)col * kInP;
    const __bf16* wob = wop + (size_t)col * kInP;

    const int srow = t >> 2;                   // staging: 16 rows x 4 segs
    const int sseg = t & 3;
    const __bf16* sbase = embs_bf + (size_t)(row0 + srow) * kInP + sseg * 8;
    __bf16* sdst = &As[0][0] + srow * 32 + sseg * 8;

    v8f accC = {};
    v8f accO = {};

    if (t < 64) async_copy16(sbase, sdst);     // prologue: tile 0 -> buf 0
    #pragma unroll
    for (int ks = 0; ks < NS; ++ks) {
        if (t < 64) wait_async0();             // tile ks has landed
        __syncthreads();                       // publish buf[ks&1]; also
                                               // orders prior reads of the
                                               // buffer we are about to fill
        if (t < 64 && ks + 1 < NS)             // overlap: copy tile ks+1
            async_copy16(sbase + (ks + 1) * 32,
                         sdst + ((ks + 1) & 1) * (16 * 32));

        const __bf16* Ab = &As[ks & 1][0];
        const v16bf a  = load_a_frag(Ab, rl, half);
        const v16bf bc = *(const v16bf*)(wcb + ks * 32 + half * 16);
        const v16bf bo = *(const v16bf*)(wob + ks * 32 + half * 16);

        accC = __builtin_amdgcn_wmma_f32_16x16x32_bf16(
                   false, a, false, bc, (short)0, accC, false, false);
        accO = __builtin_amdgcn_wmma_f32_16x16x32_bf16(
                   false, a, false, bo, (short)0, accO, false, false);
    }

    const float bC = bcx[col];
    const float bO = box[col];
    #pragma unroll
    for (int r = 0; r < 8; ++r) {              // M = r + half*8, N = rl
        const int row = row0 + r + half * 8;
        const float c = accC[r] + bC;
        const float o = sigmoidf_(accO[r] + bO);
        cbuf[(size_t)row * kMem + col] = c;
        hbf [(size_t)row * kMem + col] = (__bf16)(o * tanhf_(c));
    }
}

// ---------------------------------------------------------------------------
// Tree-level kernel: block = 256 threads (8 waves), tile 16 pair-rows x 256
// outputs; each wave owns 32 outputs x 4 gates = 8 accumulators. K = 512.
// Parent (c,h) overwrites the LEFT child slot (power-of-two stride trick).
// ---------------------------------------------------------------------------
__global__ __launch_bounds__(256) void tlstm_node_kernel(
    const __bf16* __restrict__ wb,
    const float* __restrict__ bl, const float* __restrict__ br,
    float* cbuf, __bf16* hbf, int pshift, int stride_in)
{
    constexpr int NS = 16;                     // K = 512
    __shared__ __align__(32) __bf16 As[2][16 * 32];

    const int t     = threadIdx.x;
    const int wave  = t >> 5;
    const int lane  = t & 31;
    const int half  = lane >> 4;
    const int rl    = lane & 15;
    const int row0  = blockIdx.x * 16;
    const int pmask = (1 << pshift) - 1;

    // Staging geometry: thread t<64 owns (row = t>>2, seg = t&3). The child
    // node (left vs right h) depends only on which K-half we are in.
    const int srow  = t >> 2;
    const int sseg  = t & 3;
    const int sgrow = row0 + srow;
    const int sb    = sgrow >> pshift;
    const int sj    = sgrow & pmask;
    const size_t lbase = (size_t)sb * kL + (size_t)(2 * sj) * stride_in;
    const __bf16* slh = hbf + lbase * kMem + sseg * 8;                   // k<256
    const __bf16* srh = hbf + (lbase + stride_in) * kMem + sseg * 8;     // k>=256
    __bf16* sdst = &As[0][0] + srow * 32 + sseg * 8;

    // Per-accumulator B base pointers (gate-major columns of fused Wb).
    const __bf16* wptr[8];
    #pragma unroll
    for (int ot = 0; ot < 2; ++ot) {
        const int o = wave * 32 + ot * 16 + rl;
        #pragma unroll
        for (int g = 0; g < 4; ++g)
            wptr[ot * 4 + g] = wb + (size_t)(g * kMem + o) * 512;
    }

    v8f acc[8] = {};

    if (t < 64) async_copy16(slh, sdst);       // prologue: tile 0 -> buf 0
    #pragma unroll
    for (int ks = 0; ks < NS; ++ks) {
        if (t < 64) wait_async0();
        __syncthreads();
        if (t < 64 && ks + 1 < NS) {
            const int k0 = (ks + 1) * 32;      // chunk never crosses 256
            const __bf16* g = ((k0 < kMem) ? slh : srh) + (k0 & (kMem - 1));
            async_copy16(g, sdst + ((ks + 1) & 1) * (16 * 32));
        }

        const __bf16* Ab = &As[ks & 1][0];
        const v16bf a = load_a_frag(Ab, rl, half);
        const int koff = ks * 32 + half * 16;
        #pragma unroll
        for (int i = 0; i < 8; ++i) {
            const v16bf bbf = *(const v16bf*)(wptr[i] + koff);
            acc[i] = __builtin_amdgcn_wmma_f32_16x16x32_bf16(
                         false, a, false, bbf, (short)0, acc[i], false, false);
        }
    }

    // Epilogue: gates -> (c,h). Bias sums hoisted per o-tile; child node
    // addresses depend only on the row, so compute them once per r.
    const int o0 = wave * 32 + rl;
    const int o1 = o0 + 16;
    float bsum[2][4];
    #pragma unroll
    for (int g = 0; g < 4; ++g) {
        bsum[0][g] = bl[g * kMem + o0] + br[g * kMem + o0];
        bsum[1][g] = bl[g * kMem + o1] + br[g * kMem + o1];
    }
    #pragma unroll
    for (int r = 0; r < 8; ++r) {
        const int grow = row0 + r + half * 8;
        const int b = grow >> pshift;
        const int j = grow & pmask;
        const size_t lnode = (size_t)b * kL + (size_t)(2 * j) * stride_in;
        const size_t rnode = lnode + (size_t)stride_in;
        #pragma unroll
        for (int ot = 0; ot < 2; ++ot) {
            const int o = (ot == 0) ? o0 : o1;
            const float lc = cbuf[lnode * kMem + o];
            const float rc = cbuf[rnode * kMem + o];
            const float ig = sigmoidf_(acc[ot * 4 + 0][r] + bsum[ot][0]);
            const float lf = sigmoidf_(acc[ot * 4 + 1][r] + bsum[ot][1]);
            const float rf = sigmoidf_(acc[ot * 4 + 2][r] + bsum[ot][2]);
            const float u  = tanhf_   (acc[ot * 4 + 3][r] + bsum[ot][3]);
            const float cn = ig * u + lf * lc + rf * rc;
            cbuf[lnode * kMem + o] = cn;
            hbf [lnode * kMem + o] = (__bf16)tanhf_(cn);
        }
    }
}

// ---------------------------------------------------------------------------
// Extraction: out = stack([c_root, tanh(c_root)]) -> (2, B, MEM) f32.
// ---------------------------------------------------------------------------
__global__ __launch_bounds__(256) void tlstm_extract_kernel(
    const float* __restrict__ cbuf, float* __restrict__ out)
{
    const int i = blockIdx.x * blockDim.x + threadIdx.x;  // 0 .. B*MEM-1
    const int b = i / kMem;
    const int m = i % kMem;
    const float c = cbuf[(size_t)b * kL * kMem + m];       // root = node 0 slot
    out[i]             = c;
    out[kB * kMem + i] = tanhf_(c);
}

extern "C" void kernel_launch(void* const* d_in, const int* in_sizes, int n_in,
                              void* d_out, int out_size, void* d_ws, size_t ws_size,
                              hipStream_t stream) {
    const float* embs = (const float*)d_in[0];
    const float* Wcx  = (const float*)d_in[1];
    const float* bcx  = (const float*)d_in[2];
    const float* Wox  = (const float*)d_in[3];
    const float* box  = (const float*)d_in[4];
    const float* Wl   = (const float*)d_in[5];
    const float* bl   = (const float*)d_in[6];
    const float* Wr   = (const float*)d_in[7];
    const float* br   = (const float*)d_in[8];

    // Workspace layout:
    //   cbuf   : f32  (B,L,MEM)   64 MB   | hbf : bf16 (B,L,MEM)  32 MB
    //   embsbf : bf16 (B*L,320)   40 MB   | wcp/wop: bf16 256x320 2x160 KB
    //   wb     : bf16 (4*256,512)  1 MB
    char* p = (char*)d_ws;
    float*  cbuf   = (float*)p;           p += (size_t)kB * kL * kMem * 4;
    __bf16* hbf    = (__bf16*)p;          p += (size_t)kB * kL * kMem * 2;
    __bf16* embsbf = (__bf16*)p;          p += (size_t)kB * kL * kInP * 2;
    __bf16* wcp    = (__bf16*)p;          p += (size_t)kMem * kInP * 2;
    __bf16* wop    = (__bf16*)p;          p += (size_t)kMem * kInP * 2;
    __bf16* wb     = (__bf16*)p;

    conv_embs_kernel<<<(kB * kL * (kInP / 4)) / 256, 256, 0, stream>>>(embs, embsbf);
    conv_wx_kernel<<<(kMem * (kInP / 4)) / 256, 256, 0, stream>>>(Wcx, Wox, wcp, wop);
    conv_wlr_kernel<<<(4 * kMem * 128) / 256, 256, 0, stream>>>(Wl, Wr, wb);

    dim3 lgrid((kB * kL) / 16, kMem / 128);
    tlstm_leaf_kernel<<<lgrid, 256, 0, stream>>>(embsbf, wcp, wop, bcx, box,
                                                 cbuf, hbf);

    for (int lvl = 0; lvl < 10; ++lvl) {
        const int pshift    = 9 - lvl;
        const int stride_in = 1 << lvl;
        const int rows      = kB << pshift;
        tlstm_node_kernel<<<rows / 16, 256, 0, stream>>>(wb, bl, br,
                                                         cbuf, hbf,
                                                         pshift, stride_in);
    }

    tlstm_extract_kernel<<<(kB * kMem) / 256, 256, 0, stream>>>(cbuf,
                                                                (float*)d_out);
}
